// BasicBlock_80522046865670
// MI455X (gfx1250) — compile-verified, matrix operand order checked by reading, not run on hardware
//
#include <hip/hip_runtime.h>

// ============================================================================
// MI455X / gfx1250 implementation of the ISTA-Net BasicBlock reference.
// - All GEMM-shaped work (Phi projections, 1x1 convs, channel-gram attention,
//   3x3 conv taps) lowers to v_wmma_f32_16x16x32_f16 (f16 in, f32 accumulate).
// - 1x1-conv weights are staged global->LDS via the Tensor Data Mover
//   (tensor_load_to_lds + s_wait_tensorcnt), then read as ds_load_b128.
// - q/k are pre-packed to f16 planes so the K=65536 attention-gram loop is
//   2x global_load_b128 per fragment (no per-element cvt in the hot loop).
// One wave (32 threads) per 16x16 output tile; EXEC all ones for WMMA.
//
// d_in flattening assumption (setup_inputs insertion order, params depth-first):
//  0:x 1:pre 2:pre2 3:PhiWeight 4:PhiTb 5:x_record 6:g_record
//  7:lambda_step 8:lb_max 9:lb_min 10:thres
//  11..27: atten1 {ln1_w,ln1_b,ln2_w,ln2_b,q1_w,q1_b,qdw_w,qdw_b,q_eca,
//                  kv1_w,kv1_b,kvdw_w,kvdw_b,kv_eca,out_w,out_b,out_eca}
//  28..44: atten2 (same order)   45:pre_w 46:pre_b 47:post_w 48:post_b
//  last (n_in-1): i
// Workspace requirement: ~475 MB (7 full (B,32,H,W) f32 planes + Phi scratch).
// ============================================================================

typedef __attribute__((ext_vector_type(16))) _Float16 v16h;
typedef __attribute__((ext_vector_type(8)))  float    v8f;
typedef __attribute__((ext_vector_type(4)))  unsigned int u32x4;
typedef __attribute__((ext_vector_type(4)))  int      i32x4;
typedef __attribute__((ext_vector_type(8)))  int      i32x8;

namespace {
constexpr int Hh   = 256;
constexpr int Wd   = 256;
constexpr int NPIX = Hh * Wd;          // 65536
constexpr int Bn   = 8;
constexpr int PLANE = Bn * 32 * NPIX;  // 16,777,216 floats
constexpr int BHW   = Bn * NPIX;       // 524,288 floats
}

#define WMMA32(acc, a, b) \
  (acc) = __builtin_amdgcn_wmma_f32_16x16x32_f16(false, (a), false, (b), \
                                                 (short)0, (acc), false, false)

// ---------------------------------------------------------------------------
// TDM: DMA a rows x cols f32 tile (row stride = cols elements) into LDS.
// D# layout per CDNA5 ISA ch.8: group0 = {flags, lds_addr, global_addr, type},
// group1 = {mask/data_size, tensor dims, tile dims, strides}. 2D tensor ->
// groups 2/3 NULL (zero-filled operands). Completion via TENSORcnt.
// ---------------------------------------------------------------------------
__device__ __forceinline__ void tdm_load_tile_f32(const float* gptr,
                                                  void* lds_ptr,
                                                  int cols, int rows) {
  unsigned long long ga = (unsigned long long)(const void*)gptr;
  unsigned lds = (unsigned)(unsigned long long)lds_ptr;
  u32x4 g0;
  g0[0] = 1u;                                        // count=1, user descriptor
  g0[1] = lds;                                       // lds_addr (bytes)
  g0[2] = (unsigned)(ga & 0xffffffffu);              // global_addr[31:0]
  g0[3] = (unsigned)((ga >> 32) & 0x01ffffffu) | (2u << 30);  // addr[56:32],type=2
  i32x8 g1;
  g1[0] = (2 << 16);                                 // data_size = 4 bytes
  g1[1] = (cols & 0xffff) << 16;                     // tensor_dim0 lo16
  g1[2] = (rows & 0xffff) << 16;                     // dim0 hi=0 | tensor_dim1 lo16
  g1[3] = (cols & 0xffff) << 16;                     // dim1 hi=0 | tile_dim0
  g1[4] = (rows & 0xffff);                           // tile_dim1 | tile_dim2=0
  g1[5] = cols;                                      // tensor_dim0_stride lo32
  g1[6] = 0;                                         // stride0 hi | stride1 lo
  g1[7] = 0;                                         // stride1 hi
  i32x4 z4 = {0, 0, 0, 0};
#if defined(__clang_major__) && (__clang_major__ >= 23)
  i32x8 z8 = {0, 0, 0, 0, 0, 0, 0, 0};
  __builtin_amdgcn_tensor_load_to_lds(g0, g1, z4, z4, z8, 0);
#else
  __builtin_amdgcn_tensor_load_to_lds(g0, g1, z4, z4, 0);
#endif
  __builtin_amdgcn_s_wait_tensorcnt(0);
}

// ---------------------------------------------------------------------------
// 1) Per-batch dot(dx,dg) and sum(dx^2) for the Barzilai-Borwein lambda.
// ---------------------------------------------------------------------------
__global__ void k_dots(const float* __restrict__ x,
                       const float* __restrict__ x_record,
                       const float* __restrict__ g_record,
                       const int* __restrict__ i_ptr,
                       float* __restrict__ wss) {
  int b = blockIdx.x;
  int i = *i_ptr;
  int ip = (i > 0) ? (i - 1) : 0;
  const float* xr = x_record + ip * BHW;
  const float* g1 = g_record + i  * BHW;
  const float* g0 = g_record + ip * BHW;
  float dot = 0.f, dx2 = 0.f;
  for (int t = threadIdx.x; t < NPIX; t += blockDim.x) {
    float dx = x[b * NPIX + t] - xr[b * NPIX + t];
    float dg = g1[b * NPIX + t] - g0[b * NPIX + t];
    dot += dx * dg;
    dx2 += dx * dx;
  }
  __shared__ float s1[256], s2[256];
  s1[threadIdx.x] = dot; s2[threadIdx.x] = dx2;
  __syncthreads();
  for (int s = 128; s > 0; s >>= 1) {
    if (threadIdx.x < s) {
      s1[threadIdx.x] += s1[threadIdx.x + s];
      s2[threadIdx.x] += s2[threadIdx.x + s];
    }
    __syncthreads();
  }
  if (threadIdx.x == 0) { wss[16 + b] = s1[0]; wss[32 + b] = s2[0]; }
}

__global__ void k_lambda(const float* __restrict__ lambda_step,
                         const float* __restrict__ lb_max,
                         const float* __restrict__ lb_min,
                         const int* __restrict__ i_ptr,
                         float* __restrict__ wss) {
  if (threadIdx.x != 0) return;
  int i = *i_ptr;
  float lam;
  if (i == 0) {
    lam = *lambda_step;
  } else {
    int idx = 0; float mn = wss[16];
    for (int b = 1; b < Bn; b++)
      if (wss[16 + b] < mn) { mn = wss[16 + b]; idx = b; }
    float res = wss[32 + idx] / mn;
    float lo = *lb_min, hi = *lb_max;
    lam = (mn > 0.f) ? fminf(fmaxf(res, lo), hi) : hi;
  }
  wss[0] = lam;
}

// ---------------------------------------------------------------------------
// 2) Phi forward: t[p,m] = sum_k patch(x)[p,k] * Phi[m,k];  P=512,M=256,K=1024
// ---------------------------------------------------------------------------
__global__ void k_phi_fwd(const float* __restrict__ x,
                          const float* __restrict__ phi,
                          float* __restrict__ tpatch) {
  int tp = blockIdx.x, tm = blockIdx.y;
  int lane = threadIdx.x, rh = lane >> 4, rl = lane & 15;
  int p = tp * 16 + rl;
  int b = p >> 6, pq = p & 63, py = pq >> 3, px = pq & 7;
  const float* xb = x + b * NPIX + (py * 32) * Wd + px * 32;
  int m = tm * 16 + rl;
  v8f acc = {};
  for (int k0 = 0; k0 < 1024; k0 += 32) {
    v16h a, bm;
#pragma unroll
    for (int j = 0; j < 16; j++) {
      int k = k0 + rh * 16 + j;
      a[j]  = (_Float16)xb[(k >> 5) * Wd + (k & 31)];
      bm[j] = (_Float16)phi[m * 1024 + k];
    }
    WMMA32(acc, a, bm);
  }
#pragma unroll
  for (int r = 0; r < 8; r++) {
    int mr = r + 8 * rh;
    tpatch[(tp * 16 + mr) * 256 + tm * 16 + rl] = acc[r];
  }
}

// ---------------------------------------------------------------------------
// 3) Phi transpose: t2[p,k] = sum_m t[p,m]*Phi[m,k]; scatter to (B,1,H,W)
// ---------------------------------------------------------------------------
__global__ void k_phi_bwd(const float* __restrict__ tpatch,
                          const float* __restrict__ phi,
                          float* __restrict__ tfull) {
  int tp = blockIdx.x, tk = blockIdx.y;
  int lane = threadIdx.x, rh = lane >> 4, rl = lane & 15;
  int p = tp * 16 + rl;
  int kcol = tk * 16 + rl;
  v8f acc = {};
  for (int m0 = 0; m0 < 256; m0 += 32) {
    v16h a, bm;
#pragma unroll
    for (int j = 0; j < 16; j++) {
      int m = m0 + rh * 16 + j;
      a[j]  = (_Float16)tpatch[p * 256 + m];
      bm[j] = (_Float16)phi[m * 1024 + kcol];
    }
    WMMA32(acc, a, bm);
  }
#pragma unroll
  for (int r = 0; r < 8; r++) {
    int pp = tp * 16 + r + 8 * rh, kk = tk * 16 + rl;
    int b = pp >> 6, pq = pp & 63, py = pq >> 3, px = pq & 7;
    tfull[b * NPIX + (py * 32 + (kk >> 5)) * Wd + px * 32 + (kk & 31)] = acc[r];
  }
}

// ---------------------------------------------------------------------------
// 4) x_g = x - lam*t + lam*PhiTb ;  g = t - PhiTb
// ---------------------------------------------------------------------------
__global__ void k_update(const float* __restrict__ x,
                         const float* __restrict__ tfull,
                         const float* __restrict__ phitb,
                         const float* __restrict__ wss,
                         float* __restrict__ xg, float* __restrict__ gout) {
  int t = blockIdx.x * blockDim.x + threadIdx.x;
  if (t >= BHW) return;
  float lam = wss[0];
  float tv = tfull[t];
  xg[t]   = x[t] - lam * tv + lam * phitb[t];
  gout[t] = tv - phitb[t];
}

// ---------------------------------------------------------------------------
// 5) 1->32 channel 3x3 conv (pre_w), weights staged in LDS
// ---------------------------------------------------------------------------
__global__ void k_conv3x3_c1(const float* __restrict__ in,
                             const float* __restrict__ w,
                             const float* __restrict__ bias,
                             float* __restrict__ out) {
  __shared__ float sw[288], sb[32];
  for (int t = threadIdx.x; t < 288; t += blockDim.x) sw[t] = w[t];
  for (int t = threadIdx.x; t < 32;  t += blockDim.x) sb[t] = bias[t];
  __syncthreads();
  int t = blockIdx.x * blockDim.x + threadIdx.x;
  if (t >= BHW) return;
  int b = t / NPIX, p = t % NPIX, y = p / Wd, xq = p % Wd;
  float v[9];
#pragma unroll
  for (int ky = 0; ky < 3; ky++)
#pragma unroll
    for (int kx = 0; kx < 3; kx++) {
      int yy = y + ky - 1, xx = xq + kx - 1;
      v[ky * 3 + kx] = (yy >= 0 && yy < Hh && xx >= 0 && xx < Wd)
                         ? in[b * NPIX + yy * Wd + xx] : 0.f;
    }
  for (int c = 0; c < 32; c++) {
    float acc = sb[c];
#pragma unroll
    for (int q = 0; q < 9; q++) acc += sw[c * 9 + q] * v[q];
    out[(b * 32 + c) * NPIX + p] = acc;
  }
}

// ---------------------------------------------------------------------------
// 6) LayerNorm over the 32-channel dim, per pixel
// ---------------------------------------------------------------------------
__global__ void k_ln(const float* __restrict__ in,
                     const float* __restrict__ w,
                     const float* __restrict__ b,
                     float* __restrict__ out) {
  int t = blockIdx.x * blockDim.x + threadIdx.x;
  if (t >= BHW) return;
  int bb = t / NPIX, p = t % NPIX;
  const float* base = in + (bb * 32) * NPIX + p;
  float s = 0.f, s2 = 0.f;
  for (int c = 0; c < 32; c++) { float v = base[c * NPIX]; s += v; s2 += v * v; }
  float mu = s * (1.f / 32.f);
  float var = s2 * (1.f / 32.f) - mu * mu;
  float inv = rsqrtf(var + 1e-5f);
  float* ob = out + (bb * 32) * NPIX + p;
  for (int c = 0; c < 32; c++)
    ob[c * NPIX] = (base[c * NPIX] - mu) * inv * w[c] + b[c];
}

// ---------------------------------------------------------------------------
// 7) 1x1 conv / att@v as WMMA GEMM.  out[b,co,pix] = sum_ci A[co,ci]*in[b,ci,pix]
//    K = Cin = 32 -> exactly one v_wmma per 16x16 tile.  The 32x32 weight
//    block is DMA'd into LDS by the Tensor Data Mover.
// ---------------------------------------------------------------------------
__global__ void k_gemm1x1(const float* __restrict__ A,
                          const float* __restrict__ bias,
                          const float* __restrict__ in,
                          float* __restrict__ out, int a_per_batch) {
  __shared__ float sA[32 * 32];
  int tn = blockIdx.x, tm = blockIdx.y, b = blockIdx.z;
  int lane = threadIdx.x, rh = lane >> 4, rl = lane & 15;
  const float* Ab = A + (a_per_batch ? b * 32 * 32 : 0);
  tdm_load_tile_f32(Ab, sA, 32, 32);        // TDM: 32x32 f32 tile -> LDS
  int co  = tm * 16 + rl;
  int pix = tn * 16 + rl;
  v16h a, bm;
#pragma unroll
  for (int j = 0; j < 16; j++) {
    int ci = rh * 16 + j;
    a[j]  = (_Float16)sA[co * 32 + ci];
    bm[j] = (_Float16)in[(b * 32 + ci) * NPIX + pix];
  }
  v8f acc = {};
  WMMA32(acc, a, bm);
#pragma unroll
  for (int r = 0; r < 8; r++) {
    int mr = tm * 16 + r + 8 * rh;
    float bv = bias ? bias[mr] : 0.f;
    out[(b * 32 + mr) * NPIX + tn * 16 + rl] = acc[r] + bv;
  }
}

// ---------------------------------------------------------------------------
// 8) Depthwise 3x3 (32 planes/batch; weight/bias pointers pre-offset by caller)
// ---------------------------------------------------------------------------
__global__ void k_dwconv(const float* __restrict__ in,
                         const float* __restrict__ w,
                         const float* __restrict__ bias,
                         float* __restrict__ out) {
  int t = blockIdx.x * blockDim.x + threadIdx.x;
  if (t >= PLANE) return;
  int p = t % NPIX, bc = t / NPIX, c = bc % 32;
  int y = p / Wd, xq = p % Wd;
  float acc = bias[c];
#pragma unroll
  for (int ky = 0; ky < 3; ky++)
#pragma unroll
    for (int kx = 0; kx < 3; kx++) {
      int yy = y + ky - 1, xx = xq + kx - 1;
      if (yy >= 0 && yy < Hh && xx >= 0 && xx < Wd)
        acc += w[c * 9 + ky * 3 + kx] * in[bc * NPIX + yy * Wd + xx];
    }
  out[t] = acc;
}

// ---------------------------------------------------------------------------
// 9) Plane reduction: mode 0 = mean (ECA pool), mode 1 = L2 norm (clamped)
// ---------------------------------------------------------------------------
__global__ void k_rowreduce(const float* __restrict__ in,
                            float* __restrict__ y,
                            int coff, int ystride, int mode) {
  int bc = blockIdx.x;                 // (B*32) planes
  int b = bc / 32, c = bc % 32;
  const float* base = in + bc * NPIX;
  float s = 0.f;
  for (int t = threadIdx.x; t < NPIX; t += blockDim.x) {
    float v = base[t];
    s += mode ? v * v : v;
  }
  __shared__ float sh[256];
  sh[threadIdx.x] = s;
  __syncthreads();
  for (int st = 128; st > 0; st >>= 1) {
    if (threadIdx.x < st) sh[threadIdx.x] += sh[threadIdx.x + st];
    __syncthreads();
  }
  if (threadIdx.x == 0) {
    float r = sh[0];
    r = mode ? fmaxf(sqrtf(r), 1e-12f) : r * (1.f / NPIX);
    y[b * ystride + coff + c] = r;
  }
}

// 10) ECA: 1d conv (k=3, pad 1) over channels + sigmoid
__global__ void k_eca(const float* __restrict__ y, const float* __restrict__ wk,
                      float* __restrict__ s, int C) {
  int t = blockIdx.x * blockDim.x + threadIdx.x;
  if (t >= Bn * C) return;
  int b = t / C, c = t % C;
  float acc = 0.f;
#pragma unroll
  for (int j = 0; j < 3; j++) {
    int cc = c + j - 1;
    if (cc >= 0 && cc < C) acc += wk[j] * y[b * C + cc];
  }
  s[t] = 1.f / (1.f + __expf(-acc));
}

// 11) x *= sigmoid-scale (optional residual, may be in-place)
__global__ void k_scale(const float* __restrict__ in, const float* __restrict__ s,
                        const float* __restrict__ resid, float* __restrict__ out,
                        int Ctot, int coff) {
  int t = blockIdx.x * blockDim.x + threadIdx.x;
  if (t >= PLANE) return;
  int bc = t / NPIX, b = bc / 32, c = bc % 32;
  float v = in[t] * s[b * Ctot + coff + c];
  if (resid) v += resid[t];
  out[t] = v;
}

// 11b) pack an f32 plane-set to f16 (feeds the long-K gram WMMA loop)
__global__ void k_cvt16(const float* __restrict__ in,
                        _Float16* __restrict__ out) {
  int t = blockIdx.x * blockDim.x + threadIdx.x;
  if (t >= PLANE) return;
  out[t] = (_Float16)in[t];
}

// ---------------------------------------------------------------------------
// 12) att[b,c,d] = (q[c,:].k[d,:]) / (|q_c||k_d|), K = 65536 via WMMA loop.
//     q,k are pre-packed f16; each fragment is a direct aligned v16h load
//     (2x global_load_b128 per operand per step).
// ---------------------------------------------------------------------------
__global__ void k_attqk(const _Float16* __restrict__ q,
                        const _Float16* __restrict__ k,
                        const float* __restrict__ qn,
                        const float* __restrict__ kn,
                        float* __restrict__ att) {
  int tc = blockIdx.x, td = blockIdx.y, b = blockIdx.z;
  int lane = threadIdx.x, rh = lane >> 4, rl = lane & 15;
  const _Float16* qb = q + (b * 32 + tc * 16 + rl) * NPIX;
  const _Float16* kb = k + (b * 32 + td * 16 + rl) * NPIX;
  v8f acc = {};
  for (int n0 = 0; n0 < NPIX; n0 += 32) {
    if (n0 + 4096 < NPIX) {           // uniform condition, no lane divergence
      __builtin_prefetch(qb + n0 + 4096);
      __builtin_prefetch(kb + n0 + 4096);
    }
    v16h a  = *(const v16h*)(qb + n0 + rh * 16);
    v16h bm = *(const v16h*)(kb + n0 + rh * 16);
    WMMA32(acc, a, bm);
  }
#pragma unroll
  for (int r = 0; r < 8; r++) {
    int c = tc * 16 + r + 8 * rh, d = td * 16 + rl;
    att[b * 1024 + c * 32 + d] = acc[r] / (qn[b * 32 + c] * kn[b * 32 + d]);
  }
}

// 13) Row softmax over 32 (B*32 rows)
__global__ void k_softmax32(float* __restrict__ att) {
  int t = blockIdx.x * blockDim.x + threadIdx.x;
  if (t >= Bn * 32) return;
  float* row = att + t * 32;
  float mx = row[0];
  for (int j = 1; j < 32; j++) mx = fmaxf(mx, row[j]);
  float s = 0.f;
  for (int j = 0; j < 32; j++) { float e = __expf(row[j] - mx); row[j] = e; s += e; }
  float inv = 1.f / s;
  for (int j = 0; j < 32; j++) row[j] *= inv;
}

// 14) soft threshold: sign(v)*relu(|v|-thres), v = x_mid + broadcast(x_g)
__global__ void k_softthresh(const float* __restrict__ xmid,
                             const float* __restrict__ xg,
                             const float* __restrict__ thres,
                             float* __restrict__ out) {
  int t = blockIdx.x * blockDim.x + threadIdx.x;
  if (t >= PLANE) return;
  int b = t / (32 * NPIX), p = t % NPIX;
  float v = xmid[t] + xg[b * NPIX + p];
  float a = fabsf(v) - *thres;
  float sg = (v > 0.f) ? 1.f : ((v < 0.f) ? -1.f : 0.f);
  out[t] = sg * fmaxf(a, 0.f);
}

// 15) concat([x_g, pre2], axis=1)
__global__ void k_concat(const float* __restrict__ xg,
                         const float* __restrict__ pre2,
                         float* __restrict__ out) {
  int t = blockIdx.x * blockDim.x + threadIdx.x;
  if (t >= PLANE) return;
  int bc = t / NPIX, p = t % NPIX, b = bc / 32, c = bc % 32;
  out[t] = (c == 0) ? xg[b * NPIX + p] : pre2[(b * 31 + (c - 1)) * NPIX + p];
}

// ---------------------------------------------------------------------------
// 16) post 3x3 32->32 conv as 9 accumulated WMMA taps + bias + residual
// ---------------------------------------------------------------------------
__global__ void k_conv3x3_wmma(const float* __restrict__ Wt,
                               const float* __restrict__ bias,
                               const float* __restrict__ in,
                               const float* __restrict__ resid,
                               float* __restrict__ out) {
  int tn = blockIdx.x, tm = blockIdx.y, b = blockIdx.z;
  int lane = threadIdx.x, rh = lane >> 4, rl = lane & 15;
  int pix0 = tn * 16;                 // 16 pixels of one image row (16 | Wd)
  int y = pix0 / Wd, x0 = pix0 % Wd;
  int co = tm * 16 + rl;
  v8f acc = {};
  for (int ky = 0; ky < 3; ky++) {
    int yy = y + ky - 1;
    for (int kx = 0; kx < 3; kx++) {
      int xx = x0 + rl + kx - 1;
      bool inb = (yy >= 0 && yy < Hh && xx >= 0 && xx < Wd);
      v16h a, bm;
#pragma unroll
      for (int j = 0; j < 16; j++) {
        int ci = rh * 16 + j;
        a[j]  = (_Float16)Wt[((co * 32 + ci) * 3 + ky) * 3 + kx];
        bm[j] = (_Float16)(inb ? in[(b * 32 + ci) * NPIX + yy * Wd + xx] : 0.f);
      }
      WMMA32(acc, a, bm);
    }
  }
#pragma unroll
  for (int r = 0; r < 8; r++) {
    int mr = tm * 16 + r + 8 * rh;
    int idx = (b * 32 + mr) * NPIX + pix0 + rl;
    out[idx] = acc[r] + bias[mr] + resid[idx];
  }
}

// ===========================================================================
// Host-side attention sequence.  pb = first param leaf index of this atten.
// Scratch planes P1..P5 each hold (B,32,H,W) f32.
// ===========================================================================
static void run_atten(hipStream_t stream, void* const* d_in, int pb,
                      const float* preP, const float* curP, float* outP,
                      float* P1, float* P2, float* P3, float* P4, float* P5,
                      float* Y, float* SSC, float* QN, float* KN, float* ATT) {
  auto L = [&](int k) { return (const float*)d_in[pb + k]; };
  dim3 eg((PLANE + 255) / 256), pg((BHW + 255) / 256);
  dim3 gtile(NPIX / 16, 2, Bn);

  // q path: LN(cur) -> 1x1 -> dw3x3 -> ECA
  k_ln<<<pg, 256, 0, stream>>>(curP, L(2), L(3), P1);
  k_gemm1x1<<<gtile, 32, 0, stream>>>(L(4), L(5), P1, P2, 0);
  k_dwconv<<<eg, 256, 0, stream>>>(P2, L(6), L(7), P3);
  k_rowreduce<<<256, 256, 0, stream>>>(P3, Y, 0, 32, 0);
  k_eca<<<1, 256, 0, stream>>>(Y, L(8), SSC, 32);
  k_scale<<<eg, 256, 0, stream>>>(P3, SSC, nullptr, P3, 32, 0);   // q = P3

  // kv path: LN(pre) -> split 1x1 -> split dw3x3 -> joint ECA over 64 ch
  k_ln<<<pg, 256, 0, stream>>>(preP, L(0), L(1), P1);
  k_gemm1x1<<<gtile, 32, 0, stream>>>(L(9), L(10), P1, P2, 0);               // k half
  k_gemm1x1<<<gtile, 32, 0, stream>>>(L(9) + 1024, L(10) + 32, P1, P4, 0);   // v half
  k_dwconv<<<eg, 256, 0, stream>>>(P2, L(11), L(12), P5);                    // k -> P5
  k_dwconv<<<eg, 256, 0, stream>>>(P4, L(11) + 288, L(12) + 32, P1);         // v -> P1
  k_rowreduce<<<256, 256, 0, stream>>>(P5, Y, 0, 64, 0);
  k_rowreduce<<<256, 256, 0, stream>>>(P1, Y, 32, 64, 0);
  k_eca<<<2, 256, 0, stream>>>(Y, L(13), SSC, 64);
  k_scale<<<eg, 256, 0, stream>>>(P5, SSC, nullptr, P5, 64, 0);   // k = P5
  k_scale<<<eg, 256, 0, stream>>>(P1, SSC, nullptr, P1, 64, 32);  // v = P1

  // channel-gram attention (f32 norms, f16 operands for the K=65536 loop)
  k_rowreduce<<<256, 256, 0, stream>>>(P3, QN, 0, 32, 1);
  k_rowreduce<<<256, 256, 0, stream>>>(P5, KN, 0, 32, 1);
  _Float16* q16 = (_Float16*)P2;      // P2/P4 free here; reused as f16 packs
  _Float16* k16 = (_Float16*)P4;
  k_cvt16<<<eg, 256, 0, stream>>>(P3, q16);
  k_cvt16<<<eg, 256, 0, stream>>>(P5, k16);
  k_attqk<<<dim3(2, 2, Bn), 32, 0, stream>>>(q16, k16, QN, KN, ATT);
  k_softmax32<<<1, 256, 0, stream>>>(ATT);
  k_gemm1x1<<<gtile, 32, 0, stream>>>(ATT, nullptr, P1, P2, 1);   // out = att@v

  // output 1x1 conv + ECA + residual(cur)
  k_gemm1x1<<<gtile, 32, 0, stream>>>(L(14), L(15), P2, P4, 0);
  k_rowreduce<<<256, 256, 0, stream>>>(P4, Y, 0, 32, 0);
  k_eca<<<1, 256, 0, stream>>>(Y, L(16), SSC, 32);
  k_scale<<<eg, 256, 0, stream>>>(P4, SSC, curP, outP, 32, 0);
}

// ===========================================================================
extern "C" void kernel_launch(void* const* d_in, const int* in_sizes, int n_in,
                              void* d_out, int out_size, void* d_ws, size_t ws_size,
                              hipStream_t stream) {
  (void)in_sizes; (void)out_size; (void)ws_size;
  const float* x     = (const float*)d_in[0];
  const float* pre   = (const float*)d_in[1];
  const float* pre2  = (const float*)d_in[2];
  const float* phi   = (const float*)d_in[3];
  const float* phitb = (const float*)d_in[4];
  const float* xrec  = (const float*)d_in[5];
  const float* grec  = (const float*)d_in[6];
  const int*   ip    = (const int*)d_in[n_in - 1];

  float* out    = (float*)d_out;
  float* x_pred = out;                 // (B,32,H,W)
  float* x_mid  = out + PLANE;         // (B,32,H,W)
  float* g_out  = out + 2 * PLANE;     // (B,1,H,W)

  // workspace carve-up
  float* f      = (float*)d_ws;
  float* WSS    = f;                   // 64   (lam, dots, dx2)
  float* Y      = f + 1024;            // pools
  float* SSC    = f + 2048;            // sigmoid scales
  float* QN     = f + 3072;
  float* KN     = f + 3328;
  float* ATT    = f + 4096;            // 8192
  float* TPATCH = f + 65536;           // 131072
  float* TFULL  = TPATCH + 131072;     // 524288
  float* XG     = TFULL + 524288;      // 524288
  float* P1     = XG + BHW;
  float* P2     = P1 + PLANE;
  float* P3     = P2 + PLANE;
  float* P4     = P3 + PLANE;
  float* P5     = P4 + PLANE;
  float* bufA   = P5 + PLANE;          // x_input (pre/post threshold)
  float* bufP   = bufA + PLANE;        // concat([x_g, pre2])

  dim3 eg((PLANE + 255) / 256), pg((BHW + 255) / 256);

  // lambda
  k_dots<<<Bn, 256, 0, stream>>>(x, xrec, grec, ip, WSS);
  k_lambda<<<1, 32, 0, stream>>>((const float*)d_in[7], (const float*)d_in[8],
                                 (const float*)d_in[9], ip, WSS);
  // Phi / PhiT projection
  k_phi_fwd<<<dim3(32, 16), 32, 0, stream>>>(x, phi, TPATCH);
  k_phi_bwd<<<dim3(32, 64), 32, 0, stream>>>(TPATCH, phi, TFULL);
  k_update<<<pg, 256, 0, stream>>>(x, TFULL, phitb, WSS, XG, g_out);

  // x_input = conv3x3(x_g, pre_w) ; atten1 -> x_mid
  k_conv3x3_c1<<<pg, 256, 0, stream>>>(XG, (const float*)d_in[45],
                                       (const float*)d_in[46], bufA);
  run_atten(stream, d_in, 11, pre, bufA, x_mid, P1, P2, P3, P4, P5,
            Y, SSC, QN, KN, ATT);

  // soft threshold, concat, atten2
  k_softthresh<<<eg, 256, 0, stream>>>(x_mid, XG, (const float*)d_in[10], bufA);
  k_concat<<<eg, 256, 0, stream>>>(XG, pre2, bufP);
  run_atten(stream, d_in, 28, bufA, bufP, P3, P1, P2, P3, P4, P5,
            Y, SSC, QN, KN, ATT);     // atten2 output lands in P3 (free by then)

  // x_pred = conv3x3(xo, post_w) + bias + x_input
  k_conv3x3_wmma<<<dim3(NPIX / 16, 2, Bn), 32, 0, stream>>>(
      (const float*)d_in[47], (const float*)d_in[48], P3, bufA, x_pred);
}
